// SuperPointInterestPointDecoder_53377853555276
// MI455X (gfx1250) — matrix operand order repaired and checked by reading, not compiled, not run on hardware
//
#include <hip/hip_runtime.h>
#include <stdint.h>

typedef _Float16 v16h __attribute__((ext_vector_type(16)));
typedef _Float16 v8h  __attribute__((ext_vector_type(8)));
typedef float    v8f  __attribute__((ext_vector_type(8)));
typedef int      v4i  __attribute__((ext_vector_type(4)));

#define HW     256
#define CIN    128
#define CMID   256
#define S2     2048
#define CAP    65536
#define THRESH 0.005f
#define WROW   40            // LDS row pitch (halves): 20 words -> conflict-free, 16B aligned

// ---- workspace layout (bytes) ----
#define OFF_ACT    ((size_t)0)              // act f16 (16.78MB) -> reused as suppscores f32
#define OFF_CONV   ((size_t)16777216)       // conv f16 (33.55MB) -> reused: suppmask f32 in 2nd half
#define OFF_SCORES ((size_t)50331648)
#define OFF_MASK   ((size_t)67108864)
#define OFF_WA     ((size_t)83886080)
#define OFF_WB     ((size_t)84475904)
#define OFF_CVAL   ((size_t)84516864)
#define OFF_CIDX   ((size_t)84779008)
#define OFF_CNT    ((size_t)85041152)

#define GLOBAL_AS __attribute__((address_space(1)))
#define LOCAL_AS  __attribute__((address_space(3)))

#if defined(__HIP_DEVICE_COMPILE__) && __has_builtin(__builtin_amdgcn_global_load_async_to_lds_b128)
#define HAVE_ASYNC_LDS 1
#else
#define HAVE_ASYNC_LDS 0
#endif

__device__ __forceinline__ void async_cp_b128(const _Float16* src, _Float16* dst) {
#if HAVE_ASYNC_LDS
  GLOBAL_AS v4i* gsrc = reinterpret_cast<GLOBAL_AS v4i*>(reinterpret_cast<uintptr_t>(src));
  LOCAL_AS  v4i* ldst = reinterpret_cast<LOCAL_AS  v4i*>(reinterpret_cast<uintptr_t>(dst));
  __builtin_amdgcn_global_load_async_to_lds_b128(gsrc, ldst, 0, 0);
#else
  *(v8h*)dst = *(const v8h*)src;
#endif
}

__device__ __forceinline__ void async_stage_fence() {
#if HAVE_ASYNC_LDS
#if __has_builtin(__builtin_amdgcn_s_wait_asynccnt)
  __builtin_amdgcn_s_wait_asynccnt(0);
#else
  asm volatile("s_wait_asynccnt 0" ::: "memory");
#endif
#endif
}

__device__ __forceinline__ v16h pack16(const _Float16* p0, const _Float16* p1) {
  v8h a = *(const v8h*)p0;
  v8h b = *(const v8h*)p1;
  v16h r;
#pragma unroll
  for (int i = 0; i < 8; ++i) { r[i] = a[i]; r[i + 8] = b[i]; }
  return r;
}

__device__ __forceinline__ v8f wmma_f16(v16h a, v16h b, v8f c) {
  return __builtin_amdgcn_wmma_f32_16x16x32_f16(false, a, false, b, (short)0, c, false, false);
}

// ---------------- prep kernels ----------------
__global__ __launch_bounds__(256) void zero_cnt_k(int* cnt) { *cnt = 0; }

// NCHW f32 -> HWC f16, LDS-tiled transpose (coalesced on both sides)
__global__ __launch_bounds__(256) void prep_act_k(const float* __restrict__ in,
                                                  _Float16* __restrict__ out) {
  __shared__ float tile[64][CIN + 1];      // 129-word pitch -> conflict-free
  const int pix0 = blockIdx.x * 64;
  for (int j = threadIdx.x; j < 64 * CIN; j += 256) {
    const int c = j >> 6;
    const int p = j & 63;
    tile[p][c] = in[(size_t)c * 65536 + pix0 + p];
  }
  __syncthreads();
  for (int j = threadIdx.x; j < 64 * 16; j += 256) {
    const int p  = j >> 4;
    const int cg = j & 15;
    v8h v;
#pragma unroll
    for (int e = 0; e < 8; ++e) v[e] = (_Float16)tile[p][cg * 8 + e];
    *(v8h*)(out + (size_t)(pix0 + p) * CIN + cg * 8) = v;
  }
}

// OIHW (256,128,3,3) f32 -> [tap][O][I] f16
__global__ __launch_bounds__(256) void prep_wa_k(const float* __restrict__ in,
                                                 _Float16* __restrict__ out) {
  unsigned idx = blockIdx.x * 256u + threadIdx.x;           // (k*256 + o)*128 + i
  unsigned k = idx >> 15, r = idx & 32767u, o = r >> 7, i = r & 127u;
  out[idx] = (_Float16)in[((size_t)o * 128u + i) * 9u + k];
}

// (65,256,1,1) f32 -> [80][256] f16, rows 65..79 = 0
__global__ __launch_bounds__(256) void prep_wb_k(const float* __restrict__ in,
                                                 _Float16* __restrict__ out) {
  unsigned idx = blockIdx.x * 256u + threadIdx.x;
  unsigned o = idx >> 8;
  float v = (o < 65u) ? in[idx] : 0.0f;
  out[idx] = (_Float16)v;
}

// ---------------- conv_a: implicit GEMM M=65536 N=256 K=1152 ----------------
// block = 8 waves: (m_sub 0..1) x (n_grp 0..3); wave tile = 32M x 64N (8 accums).
// Weight panel (256N x 32K f16) staged to LDS per k-step, double-buffered with
// async global->LDS copies (ASYNCcnt) overlapping the WMMA work.
__device__ __forceinline__ void stage_w(const _Float16* __restrict__ wa,
                                        _Float16* dst, int kk, int c0, int t) {
  const _Float16* src = wa + (size_t)(kk * CMID + t) * CIN + c0;  // row t, 32 halves
  _Float16* d = dst + t * WROW;
#pragma unroll
  for (int j = 0; j < 4; ++j) async_cp_b128(src + j * 8, d + j * 8);
}

__global__ __launch_bounds__(256) void conv_a_wmma_k(const _Float16* __restrict__ act,
                                                     const _Float16* __restrict__ wa,
                                                     const float* __restrict__ bias,
                                                     _Float16* __restrict__ out) {
  __shared__ _Float16 wbuf[2][CMID * WROW];    // 2 x 20KB
  const int tid  = threadIdx.x;
  const int lane = tid & 31;
  const int wave = tid >> 5;
  const int m_sub = wave & 1;
  const int n_grp = wave >> 1;
  const int m_base = blockIdx.x * 64 + m_sub * 32;   // y*256 + x0, 32-aligned
  const int y  = m_base >> 8;
  const int x0 = m_base & 255;
  const int ml  = lane & 15;
  const int hiK = lane >> 4;

  v8f accL0 = {}, accL1 = {}, accL2 = {}, accL3 = {};
  v8f accH0 = {}, accH1 = {}, accH2 = {}, accH3 = {};

  stage_w(wa, &wbuf[0][0], 0, 0, tid);
  async_stage_fence();
  __syncthreads();

  for (int step = 0; step < 36; ++step) {
    const int kk = step >> 2;
    const int c0 = (step & 3) * 32;
    if (step + 1 < 36) {                       // prefetch next panel while computing
      const int s2 = step + 1;
      stage_w(wa, &wbuf[s2 & 1][0], s2 >> 2, (s2 & 3) * 32, tid);
    }
    const int ky = kk / 3 - 1, kx = kk % 3 - 1;
    const int yi  = y + ky;
    const int xiL = x0 + ml + kx;
    const int xiH = x0 + 16 + ml + kx;
    const bool rowok = (unsigned)yi < HW;
    const bool vL = rowok && ((unsigned)xiL < HW);
    const bool vH = rowok && ((unsigned)xiH < HW);
    const int ka = c0 + hiK * 8;               // A: halves ka..ka+7 and ka+16..ka+23
    const _Float16* apL = act + (size_t)(yi * HW + xiL) * CIN + ka;
    const _Float16* apH = act + (size_t)(yi * HW + xiH) * CIN + ka;
    __builtin_prefetch(apL + CIN, 0, 1);
    v16h aL = vL ? pack16(apL, apL + 16) : (v16h)(_Float16)0.0f;
    v16h aH = vH ? pack16(apH, apH + 16) : (v16h)(_Float16)0.0f;

    const _Float16* wb0 = &wbuf[step & 1][0] + (size_t)(n_grp * 64 + ml) * WROW + hiK * 8;
    v16h b0 = pack16(wb0 + 0 * 16 * WROW, wb0 + 0 * 16 * WROW + 16);
    v16h b1 = pack16(wb0 + 1 * 16 * WROW, wb0 + 1 * 16 * WROW + 16);
    v16h b2 = pack16(wb0 + 2 * 16 * WROW, wb0 + 2 * 16 * WROW + 16);
    v16h b3 = pack16(wb0 + 3 * 16 * WROW, wb0 + 3 * 16 * WROW + 16);

    accL0 = wmma_f16(aL, b0, accL0);
    accL1 = wmma_f16(aL, b1, accL1);
    accL2 = wmma_f16(aL, b2, accL2);
    accL3 = wmma_f16(aL, b3, accL3);
    accH0 = wmma_f16(aH, b0, accH0);
    accH1 = wmma_f16(aH, b1, accH1);
    accH2 = wmma_f16(aH, b2, accH2);
    accH3 = wmma_f16(aH, b3, accH3);

    async_stage_fence();                       // staged panel complete
    __syncthreads();                           // all waves done reading current panel
  }

  const float bv0 = bias[n_grp * 64 + 0  + ml];
  const float bv1 = bias[n_grp * 64 + 16 + ml];
  const float bv2 = bias[n_grp * 64 + 32 + ml];
  const float bv3 = bias[n_grp * 64 + 48 + ml];
#pragma unroll
  for (int r = 0; r < 8; ++r) {
    const int xoL = x0 + r + hiK * 8;
    const int xoH = xoL + 16;
    _Float16* qL = out + (size_t)(y * HW + xoL) * CMID + n_grp * 64 + ml;
    _Float16* qH = out + (size_t)(y * HW + xoH) * CMID + n_grp * 64 + ml;
    qL[0]  = (_Float16)fmaxf(accL0[r] + bv0, 0.0f);
    qL[16] = (_Float16)fmaxf(accL1[r] + bv1, 0.0f);
    qL[32] = (_Float16)fmaxf(accL2[r] + bv2, 0.0f);
    qL[48] = (_Float16)fmaxf(accL3[r] + bv3, 0.0f);
    qH[0]  = (_Float16)fmaxf(accH0[r] + bv0, 0.0f);
    qH[16] = (_Float16)fmaxf(accH1[r] + bv1, 0.0f);
    qH[32] = (_Float16)fmaxf(accH2[r] + bv2, 0.0f);
    qH[48] = (_Float16)fmaxf(accH3[r] + bv3, 0.0f);
  }
}

// ---------------- conv_b + softmax + depth-to-space, M=65536 N=80(65) K=256 ----------------
__global__ __launch_bounds__(256) void conv_b_softmax_k(const _Float16* __restrict__ xin,
                                                        const _Float16* __restrict__ wb,
                                                        const float* __restrict__ bias,
                                                        float* __restrict__ scores) {
  const int lane = threadIdx.x & 31;
  const int wave = threadIdx.x >> 5;
  const int m_base = (blockIdx.x * 8 + wave) * 16;
  const int y  = m_base >> 8;
  const int x0 = m_base & 255;
  const int ml  = lane & 15;
  const int hiK = lane >> 4;

  v8f acc[5] = {v8f{}, v8f{}, v8f{}, v8f{}, v8f{}};
  const _Float16* ap = xin + (size_t)(y * HW + x0 + ml) * CMID;
#pragma unroll
  for (int c0 = 0; c0 < CMID; c0 += 32) {
    const int ka = c0 + hiK * 8;
    const int kb = c0 + hiK * 8;
    v16h a = pack16(ap + ka, ap + ka + 16);
#pragma unroll
    for (int t = 0; t < 5; ++t) {
      const _Float16* wp = wb + (size_t)(t * 16 + ml) * CMID + kb;
      v16h b = pack16(wp, wp + 16);
      acc[t] = wmma_f16(a, b, acc[t]);
    }
  }
  float bv[5];
#pragma unroll
  for (int t = 0; t < 5; ++t) {
    const int n = t * 16 + ml;
    bv[t] = (n < 65) ? bias[n] : -1.0e30f;
  }
#pragma unroll
  for (int r = 0; r < 8; ++r) {
    const int x = x0 + r + hiK * 8;
    float v[5];
    float mx = -1.0e30f;
#pragma unroll
    for (int t = 0; t < 5; ++t) { v[t] = acc[t][r] + bv[t]; mx = fmaxf(mx, v[t]); }
#pragma unroll
    for (int off = 1; off < 16; off <<= 1) mx = fmaxf(mx, __shfl_xor(mx, off, 32));
    float e[5], s = 0.0f;
#pragma unroll
    for (int t = 0; t < 5; ++t) { e[t] = __expf(v[t] - mx); s += e[t]; }
#pragma unroll
    for (int off = 1; off < 16; off <<= 1) s += __shfl_xor(s, off, 32);
    const float inv = 1.0f / s;
#pragma unroll
    for (int t = 0; t < 4; ++t) {
      const int n = t * 16 + ml;
      scores[(size_t)(y * 8 + (n >> 3)) * S2 + (x * 8 + (n & 7))] = e[t] * inv;
    }
  }
}

// ---------------- fused 9x9 max-pool (separable in LDS) + epilogues ----------------
// MODE 0: o0 = (center == pooled)             (in = scores)
// MODE 1: sup = pooled > 0; o1 = sup; o0 = sup ? 0 : aux0   (in = mask, aux0 = scores)
// MODE 2: o0 = (o0 != 0) || (center == pooled && aux0 == 0) (in = supps, aux0 = suppm, o0 = mask r/w)
template <int MODE>
__global__ __launch_bounds__(256) void pool9_k(const float* __restrict__ in,
                                               const float* __restrict__ aux0,
                                               float* __restrict__ o0,
                                               float* __restrict__ o1) {
  __shared__ float tin[24][25];
  __shared__ float th[24][17];
  const int tx = threadIdx.x & 15, ty = threadIdx.x >> 4;
  const int bx = blockIdx.x * 16, by = blockIdx.y * 16;
  for (int j = threadIdx.x; j < 24 * 24; j += 256) {
    const int r = j / 24, c = j % 24;
    const int gy = by - 4 + r, gx = bx - 4 + c;
    float v = -3.0e38f;
    if ((unsigned)gy < S2 && (unsigned)gx < S2) v = in[(size_t)gy * S2 + gx];
    tin[r][c] = v;
  }
  __syncthreads();
  for (int j = threadIdx.x; j < 24 * 16; j += 256) {
    const int r = j >> 4, c = j & 15;
    float m = tin[r][c];
#pragma unroll
    for (int d = 1; d < 9; ++d) m = fmaxf(m, tin[r][c + d]);
    th[r][c] = m;
  }
  __syncthreads();
  float p = th[ty][tx];
#pragma unroll
  for (int d = 1; d < 9; ++d) p = fmaxf(p, th[ty + d][tx]);
  const float center = tin[ty + 4][tx + 4];
  const size_t i = (size_t)(by + ty) * S2 + (bx + tx);
  if (MODE == 0) {
    o0[i] = (center == p) ? 1.0f : 0.0f;
  } else if (MODE == 1) {
    const bool sup = p > 0.0f;
    o1[i] = sup ? 1.0f : 0.0f;
    o0[i] = sup ? 0.0f : aux0[i];
  } else {
    const bool res = (o0[i] != 0.0f) || ((center == p) && (aux0[i] == 0.0f));
    o0[i] = res ? 1.0f : 0.0f;
  }
}

// ---------------- keypoint extraction ----------------
__global__ __launch_bounds__(256) void collect_k(const float* __restrict__ scores,
                                                 const float* __restrict__ mask,
                                                 float* __restrict__ cvals,
                                                 int* __restrict__ cidxs,
                                                 int* __restrict__ cnt) {
  const int x = blockIdx.x * 256 + threadIdx.x;
  const int y = blockIdx.y;
  const size_t i = (size_t)y * S2 + x;
  const float v = (mask[i] != 0.0f) ? scores[i] : 0.0f;
  if (v > THRESH && y >= 4 && x >= 4) {     // upper-bound border checks are vacuous in reference
    const int p = atomicAdd(cnt, 1);
    if (p < CAP) { cvals[p] = v; cidxs[p] = (int)i; }
  }
}

__global__ __launch_bounds__(1024) void topk_select_k(float* __restrict__ cvals,
                                                      const int* __restrict__ cidxs,
                                                      const int* __restrict__ cnt,
                                                      float* __restrict__ out) {
  __shared__ float sv[1024];
  __shared__ int   ss[1024];
  const int tid = threadIdx.x;
  int n = *cnt; if (n > CAP) n = CAP;
  for (int round = 0; round < 1024; ++round) {
    float bv = -1.0e30f; int bs = -1;
    for (int j = tid; j < n; j += 1024) {
      const float v = cvals[j];
      if (v > bv) { bv = v; bs = j; }
    }
    sv[tid] = bv; ss[tid] = bs;
    __syncthreads();
    for (int s = 512; s > 0; s >>= 1) {
      if (tid < s) {
        if (sv[tid + s] > sv[tid]) { sv[tid] = sv[tid + s]; ss[tid] = ss[tid + s]; }
      }
      __syncthreads();
    }
    if (tid == 0) {
      const int slot = ss[0];
      if (slot >= 0 && sv[0] > 0.0f) {
        const int idx = cidxs[slot];
        out[2 * round]     = (float)(idx & (S2 - 1));
        out[2 * round + 1] = (float)(idx >> 11);
        out[2048 + round]  = sv[0];
        cvals[slot] = -2.0e30f;
      } else {
        out[2 * round] = 0.0f; out[2 * round + 1] = 0.0f;
        out[2048 + round] = -1.0f;
      }
    }
    __syncthreads();
  }
}

// ---------------- launcher ----------------
extern "C" void kernel_launch(void* const* d_in, const int* in_sizes, int n_in,
                              void* d_out, int out_size, void* d_ws, size_t ws_size,
                              hipStream_t stream) {
  (void)in_sizes; (void)n_in; (void)out_size; (void)ws_size;
  const float* encoded = (const float*)d_in[0];
  const float* wA = (const float*)d_in[1];
  const float* bA = (const float*)d_in[2];
  const float* wB = (const float*)d_in[3];
  const float* bB = (const float*)d_in[4];

  char* ws = (char*)d_ws;
  _Float16* act     = (_Float16*)(ws + OFF_ACT);
  _Float16* convout = (_Float16*)(ws + OFF_CONV);
  float*    scores  = (float*)(ws + OFF_SCORES);
  float*    maskf   = (float*)(ws + OFF_MASK);
  _Float16* wa      = (_Float16*)(ws + OFF_WA);
  _Float16* wb      = (_Float16*)(ws + OFF_WB);
  float*    cvals   = (float*)(ws + OFF_CVAL);
  int*      cidxs   = (int*)(ws + OFF_CIDX);
  int*      cnt     = (int*)(ws + OFF_CNT);
  float*    suppm   = (float*)(ws + OFF_CONV + (size_t)16777216);  // reuse
  float*    supps   = (float*)(ws + OFF_ACT);                      // reuse
  float*    out     = (float*)d_out;

  zero_cnt_k<<<1, 1, 0, stream>>>(cnt);
  prep_act_k<<<1024, 256, 0, stream>>>(encoded, act);
  prep_wa_k<<<1152, 256, 0, stream>>>(wA, wa);
  prep_wb_k<<<80, 256, 0, stream>>>(wB, wb);

  conv_a_wmma_k<<<1024, 256, 0, stream>>>(act, wa, bA, convout);
  conv_b_softmax_k<<<512, 256, 0, stream>>>(convout, wb, bB, scores);

  dim3 g2(128, 128), b(256);
  pool9_k<0><<<g2, b, 0, stream>>>(scores, scores, maskf, maskf);
  for (int it = 0; it < 2; ++it) {
    pool9_k<1><<<g2, b, 0, stream>>>(maskf, scores, supps, suppm);
    pool9_k<2><<<g2, b, 0, stream>>>(supps, suppm, maskf, maskf);
  }

  dim3 gc(8, 2048);
  collect_k<<<gc, b, 0, stream>>>(scores, maskf, cvals, cidxs, cnt);
  topk_select_k<<<1, 1024, 0, stream>>>(cvals, cidxs, cnt, out);
}
